// Model_90958817395320
// MI455X (gfx1250) — matrix operand order repaired and checked by reading, not compiled, not run on hardware
//
#include <hip/hip_runtime.h>
#include <math.h>

#define DM 256
#define NHEADS 8
#define DH 32

typedef __attribute__((ext_vector_type(16))) _Float16 v16h;
typedef __attribute__((ext_vector_type(8)))  _Float16 v8h;
typedef __attribute__((ext_vector_type(4)))  _Float16 v4h;
typedef __attribute__((ext_vector_type(8)))  float    v8f;

union AFrag { v16h v; _Float16 h[16]; };
union CFrag { v8f  v; float     f[8];  };

// ---------------------------------------------------------------------------
// WMMA GEMM: C[M,N] = act(A[M,K] * W[K,N] + bias + res)
// REQUIRES: M%64==0, N%64==0, K%32==0 (true for every call in this model).
// act: 0=none, 1=exact GELU, 2=tanh
// grid: (N/64, M/64), block 256 (8 waves: 4 M-waves x 2 N-waves)
// LDS tiles are stored pre-permuted into the WMMA fragment layout so each
// lane's 16 operand halves are one contiguous 32-byte chunk (2x ds_load_b128).
// ---------------------------------------------------------------------------
__global__ __launch_bounds__(256) void gemm_wmma(
    const float* __restrict__ A, const float* __restrict__ W,
    const float* __restrict__ bias, const float* __restrict__ res,
    float* __restrict__ C, int M, int N, int K, int act)
{
  // [m-wave][lane = 16*half + m15][e]  /  [sub-tile][lane = 16*half + n15][e]
  __shared__ __align__(32) _Float16 As2[4][32][16];
  __shared__ __align__(32) _Float16 Bs2[4][32][16];

  const int tid  = threadIdx.x;
  const int wave = tid >> 5, lane = tid & 31;
  const int wm = wave >> 1;        // 0..3
  const int wn = wave & 1;         // 0..1
  const int tileM = blockIdx.y * 64;
  const int tileN = blockIdx.x * 64;
  const int half = lane >> 4;
  const int l15  = lane & 15;

  // B staging coords: each thread owns one column slice, 8 strided K rows
  const int cl  = tid & 63;        // local column 0..63
  const int kk  = tid >> 6;        // k-group 0..3 (8 rows each)
  const int bSub  = cl >> 4;       // sub-tile 0..3
  const int bLane = 16 * (kk & 1) + (cl & 15);
  const int bE0   = (kk >> 1) * 8;

  CFrag acc0, acc1;
#pragma unroll
  for (int i = 0; i < 8; i++) { acc0.f[i] = 0.f; acc1.f[i] = 0.f; }

  for (int k0 = 0; k0 < K; k0 += 32) {
    // ---- stage A (64x32): 2 float4 per thread, packed 8-byte LDS stores ----
#pragma unroll
    for (int i = 0; i < 2; i++) {
      int e4 = tid + i * 256;             // 512 float4s
      int r  = e4 >> 3;                   // row 0..63
      int c4 = (e4 & 7) * 4;              // col 0,4,...,28
      const float4 f = *(const float4*)(A + (size_t)(tileM + r) * K + k0 + c4);
      int ah   = (c4 >> 3) & 1;
      int ae   = ((c4 >> 4) << 3) | (c4 & 7);
      int alan = 16 * ah + (r & 15);
      v4h p;
      p[0] = (_Float16)f.x; p[1] = (_Float16)f.y;
      p[2] = (_Float16)f.z; p[3] = (_Float16)f.w;
      *(v4h*)&As2[r >> 4][alan][ae] = p;
    }
    // ---- stage B (32x64): 8 K-strided loads, one 16-byte LDS store ----
    {
      const float* wp = W + (size_t)(k0 + kk * 8) * N + tileN + cl;
      v8h p;
#pragma unroll
      for (int i = 0; i < 8; i++) p[i] = (_Float16)wp[(size_t)i * N];
      *(v8h*)&Bs2[bSub][bLane][bE0] = p;
    }
    __syncthreads();

    AFrag af, bf0, bf1;
    af.v  = *(const v16h*)As2[wm][lane];
    bf0.v = *(const v16h*)Bs2[wn * 2 + 0][lane];
    bf1.v = *(const v16h*)Bs2[wn * 2 + 1][lane];
    acc0.v = __builtin_amdgcn_wmma_f32_16x16x32_f16(false, af.v, false, bf0.v, (short)0, acc0.v, false, false);
    acc1.v = __builtin_amdgcn_wmma_f32_16x16x32_f16(false, af.v, false, bf1.v, (short)0, acc1.v, false, false);
    __syncthreads();
  }

#pragma unroll
  for (int r = 0; r < 8; r++) {
    int gm = tileM + wm * 16 + r + 8 * half;
#pragma unroll
    for (int s = 0; s < 2; s++) {
      int gn = tileN + wn * 32 + s * 16 + l15;
      float v = (s == 0) ? acc0.f[r] : acc1.f[r];
      if (bias) v += bias[gn];
      if (res)  v += res[(size_t)gm * N + gn];
      if (act == 1)      v = 0.5f * v * (1.f + erff(v * 0.70710678118654752f));
      else if (act == 2) v = tanhf(v);
      C[(size_t)gm * N + gn] = v;
    }
  }
}

// ---------------------------------------------------------------------------
// Threefry2x32-20 (exact JAX RNG) -> sampled key indices for prob attention
// ---------------------------------------------------------------------------
__device__ __forceinline__ unsigned rotl32(unsigned x, unsigned r) { return (x << r) | (x >> (32 - r)); }

__device__ void threefry2x32(unsigned k0, unsigned k1, unsigned x0, unsigned x1,
                             unsigned& o0, unsigned& o1)
{
  unsigned ks2 = 0x1BD11BDAu ^ k0 ^ k1;
  unsigned a = x0 + k0, b = x1 + k1;
  const unsigned RA[4] = {13, 15, 26, 6};
  const unsigned RB[4] = {17, 29, 16, 24};
#pragma unroll
  for (int i = 0; i < 4; i++) { a += b; b = rotl32(b, RA[i]); b ^= a; }
  a += k1;  b += ks2 + 1u;
#pragma unroll
  for (int i = 0; i < 4; i++) { a += b; b = rotl32(b, RB[i]); b ^= a; }
  a += ks2; b += k0 + 2u;
#pragma unroll
  for (int i = 0; i < 4; i++) { a += b; b = rotl32(b, RA[i]); b ^= a; }
  a += k0;  b += k1 + 3u;
#pragma unroll
  for (int i = 0; i < 4; i++) { a += b; b = rotl32(b, RB[i]); b ^= a; }
  a += k1;  b += ks2 + 4u;
#pragma unroll
  for (int i = 0; i < 4; i++) { a += b; b = rotl32(b, RA[i]); b ^= a; }
  a += ks2; b += k0 + 5u;
  o0 = a; o1 = b;
}

__global__ void randint_kernel(int* idx, int n, int layer, int lkmask)
{
  int j = blockIdx.x * blockDim.x + threadIdx.x;
  if (j >= n) return;
  unsigned ko0, ko1;
  threefry2x32(0u, 42u, 0u, (unsigned)layer, ko0, ko1);  // fold_in(key(42), layer)
  int h = n >> 1;
  unsigned o0, o1;
  if (j < h) { threefry2x32(ko0, ko1, (unsigned)j,       (unsigned)(j + h), o0, o1); idx[j] = (int)(o0 & (unsigned)lkmask); }
  else       { threefry2x32(ko0, ko1, (unsigned)(j - h), (unsigned)j,       o0, o1); idx[j] = (int)(o1 & (unsigned)lkmask); }
}

// ---------------------------------------------------------------------------
// Prob-sparse sparsity measure: M[b,h,l] = max_u(q.k_idx) - sum_u/L
// one wave per (b,h,l), lanes over u
// ---------------------------------------------------------------------------
__global__ __launch_bounds__(256) void prob_measure_kernel(
    const float* __restrict__ q, const float* __restrict__ k, const int* __restrict__ idx,
    float* __restrict__ Mout, int Bsz, int L, int U)
{
  int gwave = blockIdx.x * (blockDim.x >> 5) + (threadIdx.x >> 5);
  int lane = threadIdx.x & 31;
  int total = Bsz * NHEADS * L;
  if (gwave >= total) return;
  int l = gwave % L;
  int bh = gwave / L;
  int h = bh % NHEADS, b = bh / NHEADS;
  const float4* q4 = (const float4*)(q + ((size_t)(b * L + l)) * DM + h * DH);
  float4 qv[8];
#pragma unroll
  for (int j = 0; j < 8; j++) qv[j] = q4[j];
  float mx = -1e30f, sm = 0.f;
  for (int u = lane; u < U; u += 32) {
    int ks = idx[l * U + u];
    const float4* k4 = (const float4*)(k + ((size_t)(b * L + ks)) * DM + h * DH);
    float dot = 0.f;
#pragma unroll
    for (int j = 0; j < 8; j++) {
      float4 kv = k4[j];
      dot += qv[j].x * kv.x + qv[j].y * kv.y + qv[j].z * kv.z + qv[j].w * kv.w;
    }
    mx = fmaxf(mx, dot); sm += dot;
  }
  for (int o = 16; o > 0; o >>= 1) {
    mx = fmaxf(mx, __shfl_xor(mx, o, 32));
    sm += __shfl_xor(sm, o, 32);
  }
  if (lane == 0) Mout[gwave] = mx - sm / (float)L;
}

// top-u (JAX tie-break: value desc, index asc); one block per (b,h)
__global__ __launch_bounds__(256) void topk_kernel(const float* __restrict__ Mv, int* __restrict__ top,
                                                   int L, int u)
{
  __shared__ float sv[2048];
  __shared__ float rv[256];
  __shared__ int   ri[256];
  int bh = blockIdx.x, t = threadIdx.x;
  for (int i = t; i < L; i += 256) sv[i] = Mv[(size_t)bh * L + i];
  __syncthreads();
  for (int it = 0; it < u; it++) {
    float bv = -1e30f; int bi = L;
    for (int i = t; i < L; i += 256) {
      float v = sv[i];
      if (v > bv || (v == bv && i < bi)) { bv = v; bi = i; }
    }
    rv[t] = bv; ri[t] = bi;
    __syncthreads();
    for (int s = 128; s > 0; s >>= 1) {
      if (t < s) {
        if (rv[t + s] > rv[t] || (rv[t + s] == rv[t] && ri[t + s] < ri[t])) { rv[t] = rv[t + s]; ri[t] = ri[t + s]; }
      }
      __syncthreads();
    }
    if (t == 0) { top[bh * u + it] = ri[0]; sv[ri[0]] = -1e30f; }
    __syncthreads();
  }
}

__global__ void vmean_kernel(const float* __restrict__ v, float* __restrict__ vm, int L)
{
  int i = blockIdx.x * blockDim.x + threadIdx.x;  // b*256 + c
  int b = i >> 8, c = i & 255;
  float s = 0.f;
  for (int l = 0; l < L; l++) s += v[((size_t)(b * L + l)) * DM + c];
  vm[i] = s / (float)L;
}

__global__ void fillctx_kernel(const float* __restrict__ vm, float* __restrict__ out, int total, int L)
{
  int i = blockIdx.x * blockDim.x + threadIdx.x;
  if (i >= total) return;
  int c = i & 255;
  int b = (i >> 8) / L;
  out[i] = vm[b * 256 + c];
}

// ---------------------------------------------------------------------------
// Attention row kernel (shared for prob-sparse rows / full / causal)
// grid.x = B*H*nrows, block 256; out layout (B, LQ, 256) head-sliced
// ---------------------------------------------------------------------------
__global__ __launch_bounds__(256) void attn_row_kernel(
    const float* __restrict__ q, const float* __restrict__ k, const float* __restrict__ v,
    const int* __restrict__ rowsel, float* __restrict__ out,
    int nrows, int LQ, int LK, int causal, float scale)
{
  __shared__ float sc[2048];
  __shared__ float red[256];
  __shared__ float accs[8][32];
  int t = threadIdx.x;
  int i = blockIdx.x % nrows;
  int bh = blockIdx.x / nrows;
  int h = bh % NHEADS, b = bh / NHEADS;
  int row = rowsel ? rowsel[bh * nrows + i] : i;
  int limit = causal ? (row + 1) : LK;
  const float* kbase = k + ((size_t)b * LK) * DM + h * DH;
  const float* vbase = v + ((size_t)b * LK) * DM + h * DH;
  const float4* q4 = (const float4*)(q + ((size_t)(b * LQ + row)) * DM + h * DH);
  float4 qv[8];
#pragma unroll
  for (int j = 0; j < 8; j++) qv[j] = q4[j];

  float lmax = -1e30f;
  for (int s = t; s < limit; s += 256) {
    const float4* k4 = (const float4*)(kbase + (size_t)s * DM);
    float dot = 0.f;
#pragma unroll
    for (int j = 0; j < 8; j++) {
      float4 kv = k4[j];
      dot += qv[j].x * kv.x + qv[j].y * kv.y + qv[j].z * kv.z + qv[j].w * kv.w;
    }
    dot *= scale;
    sc[s] = dot;
    lmax = fmaxf(lmax, dot);
  }
  red[t] = lmax; __syncthreads();
  for (int s = 128; s > 0; s >>= 1) { if (t < s) red[t] = fmaxf(red[t], red[t + s]); __syncthreads(); }
  float gmax = red[0]; __syncthreads();

  float lsum = 0.f;
  for (int s = t; s < limit; s += 256) { float e = expf(sc[s] - gmax); sc[s] = e; lsum += e; }
  red[t] = lsum; __syncthreads();
  for (int s = 128; s > 0; s >>= 1) { if (t < s) red[t] += red[t + s]; __syncthreads(); }
  float gsum = red[0];

  int d = t & 31, g = t >> 5;
  float acc = 0.f;
  for (int s = g; s < limit; s += 8) acc += sc[s] * vbase[(size_t)s * DM + d];
  accs[g][d] = acc; __syncthreads();
  if (t < 32) {
    float a = 0.f;
#pragma unroll
    for (int gg = 0; gg < 8; gg++) a += accs[gg][d];
    out[((size_t)(b * LQ + row)) * DM + h * DH + d] = a / gsum;
  }
}

// LayerNorm over last dim 256; one wave per row
__global__ __launch_bounds__(256) void layernorm_kernel(
    const float* __restrict__ x, const float* __restrict__ g, const float* __restrict__ bta,
    float* __restrict__ out, int Mrows)
{
  int gwave = blockIdx.x * 8 + (threadIdx.x >> 5);
  int lane = threadIdx.x & 31;
  if (gwave >= Mrows) return;
  const float* xr = x + (size_t)gwave * DM;
  float vals[8]; float s = 0.f, s2 = 0.f;
#pragma unroll
  for (int j = 0; j < 8; j++) { float v = xr[j * 32 + lane]; vals[j] = v; s += v; s2 += v * v; }
  for (int o = 16; o > 0; o >>= 1) { s += __shfl_xor(s, o, 32); s2 += __shfl_xor(s2, o, 32); }
  float mean = s / 256.f;
  float var = s2 / 256.f - mean * mean;
  float rstd = rsqrtf(var + 1e-5f);
  float* orow = out + (size_t)gwave * DM;
#pragma unroll
  for (int j = 0; j < 8; j++) { int c = j * 32 + lane; orow[c] = (vals[j] - mean) * rstd * g[c] + bta[c]; }
}

// Embedding: circular conv(k=3, Cin->256) + sinusoidal pos + time proj
__global__ void embed_kernel(const float* __restrict__ x, const float* __restrict__ xmark,
                             const float* __restrict__ convw, const float* __restrict__ timew,
                             float* __restrict__ out, int L, int Cin, int total)
{
  int i = blockIdx.x * blockDim.x + threadIdx.x;
  if (i >= total) return;
  int d = i & 255; int bt = i >> 8; int t = bt % L; int b = bt / L;
  float acc = 0.f;
  for (int kk = 0; kk < 3; kk++) {
    int tt = t + kk - 1; tt = (tt < 0) ? tt + L : (tt >= L ? tt - L : tt);
    const float* xr = x + ((size_t)(b * L + tt)) * Cin;
    const float* wr = convw + ((size_t)kk * Cin) * DM + d;
    for (int c = 0; c < Cin; c++) acc += xr[c] * wr[(size_t)c * DM];
  }
  int i2 = d >> 1;
  float ang = (float)t * expf(-(float)(2 * i2) * 0.03597789207803223f); // ln(1e4)/256
  float pe = (d & 1) ? cosf(ang) : sinf(ang);
  const float* mr = xmark + ((size_t)(b * L + t)) * 4;
  float tm = 0.f;
#pragma unroll
  for (int f = 0; f < 4; f++) tm += mr[f] * timew[f * DM + d];
  out[i] = acc + pe + tm;
}

// im2col for circular conv k=3, 256 channels: cols = kk*256 + c
__global__ void im2col_kernel(const float* __restrict__ x, float* __restrict__ out, int L, int total)
{
  int i = blockIdx.x * blockDim.x + threadIdx.x;
  if (i >= total) return;
  int col = i % 768; int row = i / 768;
  int t = row % L; int b = row / L;
  int kk = col >> 8; int c = col & 255;
  int tt = t + kk - 1; tt = (tt < 0) ? tt + L : (tt >= L ? tt - L : tt);
  out[i] = x[((size_t)(b * L + tt)) * DM + c];
}

// BatchNorm + ELU + maxpool(window 3, stride 2, pad 1) over time
__global__ void bnelupool_kernel(const float* __restrict__ y,
                                 const float* __restrict__ bg, const float* __restrict__ bb,
                                 const float* __restrict__ bm, const float* __restrict__ bv,
                                 float* __restrict__ out, int L, int total)
{
  int i = blockIdx.x * blockDim.x + threadIdx.x;
  if (i >= total) return;
  int c = i & 255; int btp = i >> 8; int Lp = L >> 1;
  int tp = btp % Lp; int b = btp / Lp;
  float scale = bg[c] * rsqrtf(bv[c] + 1e-5f);
  float shift = bb[c] - bm[c] * scale;
  float best = -1e30f;
  for (int w = -1; w <= 1; w++) {
    int t = 2 * tp + w;
    if (t < 0 || t >= L) continue;
    float val = y[((size_t)(b * L + t)) * DM + c] * scale + shift;
    val = (val > 0.f) ? val : (expf(val) - 1.f);
    best = fmaxf(best, val);
  }
  out[i] = best;
}

// feat = dec[:, -512:, :]
__global__ void slice_kernel(const float* __restrict__ x, float* __restrict__ out, int total)
{
  int i = blockIdx.x * blockDim.x + threadIdx.x;
  if (i >= total) return;
  int c = i & 255; int bt = i >> 8; int t = bt & 511; int b = bt >> 9;
  out[i] = x[((size_t)(b * 1024 + 512 + t)) * DM + c];
}

// DeFi uncertainty: unc = NUM_FINE / (sum exp(-||z-proto||^2) + NUM_FINE)
__global__ void unc_kernel(const float* __restrict__ z, const float* __restrict__ protof,
                           float* __restrict__ unc, int total)
{
  int i = blockIdx.x * blockDim.x + threadIdx.x;
  if (i >= total) return;
  const float* zr = z + (size_t)i * 64;
  float S = 0.f;
  for (int p = 0; p < 4; p++) {
    float d2 = 0.f;
    const float* pr = protof + p * 64;
    for (int d = 0; d < 64; d++) { float df = zr[d] - pr[d]; d2 += df * df; }
    S += expf(-d2);
  }
  unc[i] = 4.f / (S + 4.f);
}

// out = (feat + unc*offsets) @ head_w + head_b, N=7
__global__ void head_kernel(const float* __restrict__ feat, const float* __restrict__ uncv,
                            const float* __restrict__ offs, const float* __restrict__ w,
                            const float* __restrict__ bias, float* __restrict__ out, int total)
{
  int i = blockIdx.x * blockDim.x + threadIdx.x;
  if (i >= total) return;
  int j = i % 7; int bt = i / 7;
  const float* fr = feat + (size_t)bt * DM;
  const float* orow = offs + (size_t)bt * DM;
  float u = uncv[bt];
  float acc = bias[j];
  for (int c = 0; c < DM; c++) acc += (fr[c] + u * orow[c]) * w[c * 7 + j];
  out[i] = acc;
}

// ---------------------------------------------------------------------------
static inline int cdiv(int a, int b) { return (a + b - 1) / b; }

extern "C" void kernel_launch(void* const* d_in, const int* in_sizes, int n_in,
                              void* d_out, int out_size, void* d_ws, size_t ws_size,
                              hipStream_t stream)
{
  (void)in_sizes; (void)n_in; (void)out_size; (void)ws_size;
  const float* x_enc  = (const float*)d_in[0];
  const float* xm_enc = (const float*)d_in[1];
  const float* x_dec  = (const float*)d_in[2];
  const float* xm_dec = (const float*)d_in[3];
  auto P = [&](int i) { return (const float*)d_in[i]; };
  // pytree (sorted-key) param indices:
  // conv_layers[0]: b=4 bn_b=5 bn_g=6 bn_m=7 bn_v=8 w=9
  // dec_emb: conv_w=10 time_w=11
  // dec_layers[0]: cross{k_b=12 k_w=13 o_b=14 o_w=15 q_b=16 q_w=17 v_b=18 v_w=19}
  //   ff1_b=20 ff1_w=21 ff2_b=22 ff2_w=23 ln1_b=24 ln1_g=25 ln2_b=26 ln2_g=27 ln3_b=28 ln3_g=29
  //   self{k_b=30 k_w=31 o_b=32 o_w=33 q_b=34 q_w=35 v_b=36 v_w=37}
  // dec_norm: b=38 g=39 ; defi: off_b=40 off_w=41 proj_b=42 proj_w=43 proto_c=44 proto_f=45
  // enc_emb: conv_w=46 time_w=47
  // enc_layers[i] base=48+16*i: ff1_b+0 ff1_w+1 ff2_b+2 ff2_w+3 k_b+4 k_w+5 ln1_b+6 ln1_g+7
  //   ln2_b+8 ln2_g+9 o_b+10 o_w+11 q_b+12 q_w+13 v_b+14 v_w+15
  // enc_norm: b=80 g=81 ; head: b=82 w=83

  const int B = 4;
  float* ws = (float*)d_ws; size_t off = 0;
  auto alloc = [&](size_t n) { float* p = ws + off; off += n; return p; };
  float* X    = alloc((size_t)8192 * 256);
  float* Abuf = alloc((size_t)8192 * 256);
  float* Q    = alloc((size_t)8192 * 256);
  float* Kb   = alloc((size_t)8192 * 256);
  float* Vb   = alloc((size_t)8192 * 256);
  float* Hh   = alloc((size_t)8192 * 1024);
  float* T    = alloc((size_t)8192 * 256);
  float* ENC  = alloc((size_t)4096 * 256);
  float* Mm   = alloc(65536);
  float* VM   = alloc(1024);
  float* Z    = alloc((size_t)2048 * 64);
  float* UNC  = alloc(2048);
  float* OFFb = alloc((size_t)2048 * 256);
  float* FEAT = alloc((size_t)2048 * 256);
  int* TOP = (int*)alloc(4096);
  int* IDX = (int*)alloc(131072);

  const float scale = 0.17677669529663687f; // 1/sqrt(32)

  // ================= encoder =================
  {
    int L = 2048, total = B * L * 256;
    embed_kernel<<<cdiv(total, 256), 256, 0, stream>>>(x_enc, xm_enc, P(46), P(47), X, L, 7, total);
  }
  int L = 2048;
  for (int layer = 0; layer < 2; layer++) {
    int base = 48 + 16 * layer;
    int Mr = B * L;
    int U = (layer == 0) ? 40 : 35;   // FACTOR*ceil(ln(L)), clamped
    int nIdx = L * U;
    randint_kernel<<<cdiv(nIdx, 256), 256, 0, stream>>>(IDX, nIdx, layer, L - 1);

    dim3 gqkv(4, Mr / 64);
    gemm_wmma<<<gqkv, 256, 0, stream>>>(X, P(base + 13), P(base + 12), nullptr, Q,  Mr, 256, 256, 0);
    gemm_wmma<<<gqkv, 256, 0, stream>>>(X, P(base + 5),  P(base + 4),  nullptr, Kb, Mr, 256, 256, 0);
    gemm_wmma<<<gqkv, 256, 0, stream>>>(X, P(base + 15), P(base + 14), nullptr, Vb, Mr, 256, 256, 0);

    prob_measure_kernel<<<B * NHEADS * L / 8, 256, 0, stream>>>(Q, Kb, IDX, Mm, B, L, U);
    topk_kernel<<<B * NHEADS, 256, 0, stream>>>(Mm, TOP, L, U);
    vmean_kernel<<<4, 256, 0, stream>>>(Vb, VM, L);
    fillctx_kernel<<<cdiv(Mr * 256, 256), 256, 0, stream>>>(VM, Abuf, Mr * 256, L);
    attn_row_kernel<<<B * NHEADS * U, 256, 0, stream>>>(Q, Kb, Vb, TOP, Abuf, U, L, L, 0, scale);

    gemm_wmma<<<gqkv, 256, 0, stream>>>(Abuf, P(base + 11), P(base + 10), X, T, Mr, 256, 256, 0);
    layernorm_kernel<<<Mr / 8, 256, 0, stream>>>(T, P(base + 7), P(base + 6), X, Mr);

    dim3 gff1(16, Mr / 64);
    gemm_wmma<<<gff1, 256, 0, stream>>>(X, P(base + 1), P(base + 0), nullptr, Hh, Mr, 1024, 256, 1);
    gemm_wmma<<<gqkv, 256, 0, stream>>>(Hh, P(base + 3), P(base + 2), X, T, Mr, 256, 1024, 0);
    layernorm_kernel<<<Mr / 8, 256, 0, stream>>>(T, P(base + 9), P(base + 8), X, Mr);

    if (layer == 0) {  // distilling conv block
      int tot = Mr * 768;
      im2col_kernel<<<cdiv(tot, 256), 256, 0, stream>>>(X, Hh, L, tot);
      dim3 gc(4, Mr / 64);
      gemm_wmma<<<gc, 256, 0, stream>>>(Hh, P(9), P(4), nullptr, T, Mr, 256, 768, 0);
      int Lp = L / 2; int totp = B * Lp * 256;
      bnelupool_kernel<<<cdiv(totp, 256), 256, 0, stream>>>(T, P(6), P(5), P(7), P(8), X, L, totp);
      L = Lp;
    }
  }
  layernorm_kernel<<<(B * L) / 8, 256, 0, stream>>>(X, P(81), P(80), ENC, B * L);

  // ================= decoder =================
  {
    int Ld = 1024, Mr = B * Ld, total = Mr * 256;
    embed_kernel<<<cdiv(total, 256), 256, 0, stream>>>(x_dec, xm_dec, P(10), P(11), X, Ld, 7, total);
    dim3 g256(4, Mr / 64);
    // self attention (causal)
    gemm_wmma<<<g256, 256, 0, stream>>>(X, P(35), P(34), nullptr, Q,  Mr, 256, 256, 0);
    gemm_wmma<<<g256, 256, 0, stream>>>(X, P(31), P(30), nullptr, Kb, Mr, 256, 256, 0);
    gemm_wmma<<<g256, 256, 0, stream>>>(X, P(37), P(36), nullptr, Vb, Mr, 256, 256, 0);
    attn_row_kernel<<<B * NHEADS * Ld, 256, 0, stream>>>(Q, Kb, Vb, nullptr, Abuf, Ld, Ld, Ld, 1, scale);
    gemm_wmma<<<g256, 256, 0, stream>>>(Abuf, P(33), P(32), X, T, Mr, 256, 256, 0);
    layernorm_kernel<<<Mr / 8, 256, 0, stream>>>(T, P(25), P(24), X, Mr);
    // cross attention (kv from encoder output)
    gemm_wmma<<<g256, 256, 0, stream>>>(X,   P(17), P(16), nullptr, Q,  Mr, 256, 256, 0);
    gemm_wmma<<<g256, 256, 0, stream>>>(ENC, P(13), P(12), nullptr, Kb, Mr, 256, 256, 0);
    gemm_wmma<<<g256, 256, 0, stream>>>(ENC, P(19), P(18), nullptr, Vb, Mr, 256, 256, 0);
    attn_row_kernel<<<B * NHEADS * Ld, 256, 0, stream>>>(Q, Kb, Vb, nullptr, Abuf, Ld, Ld, 1024, 0, scale);
    gemm_wmma<<<g256, 256, 0, stream>>>(Abuf, P(15), P(14), X, T, Mr, 256, 256, 0);
    layernorm_kernel<<<Mr / 8, 256, 0, stream>>>(T, P(27), P(26), X, Mr);
    // FFN
    dim3 gff(16, Mr / 64);
    gemm_wmma<<<gff, 256, 0, stream>>>(X, P(21), P(20), nullptr, Hh, Mr, 1024, 256, 1);
    gemm_wmma<<<g256, 256, 0, stream>>>(Hh, P(23), P(22), X, T, Mr, 256, 1024, 0);
    layernorm_kernel<<<Mr / 8, 256, 0, stream>>>(T, P(29), P(28), X, Mr);
    // final decoder norm
    layernorm_kernel<<<Mr / 8, 256, 0, stream>>>(X, P(39), P(38), Abuf, Mr);

    // ============ DeFi head ============
    slice_kernel<<<cdiv(2048 * 256, 256), 256, 0, stream>>>(Abuf, FEAT, 2048 * 256);
    dim3 gproj(1, 2048 / 64);
    gemm_wmma<<<gproj, 256, 0, stream>>>(FEAT, P(43), P(42), nullptr, Z, 2048, 64, 256, 0);
    unc_kernel<<<cdiv(2048, 256), 256, 0, stream>>>(Z, P(45), UNC, 2048);
    dim3 goff(4, 2048 / 64);
    gemm_wmma<<<goff, 256, 0, stream>>>(FEAT, P(41), P(40), nullptr, OFFb, 2048, 256, 256, 2);
    head_kernel<<<cdiv(2048 * 7, 256), 256, 0, stream>>>(FEAT, UNC, OFFb, P(83), P(82),
                                                         (float*)d_out, 2048 * 7);
  }
}
